// InceptionDWConv2d_70231305224429
// MI455X (gfx1250) — matrix operand order repaired and checked by reading, not compiled, not run on hardware
//
#include <hip/hip_runtime.h>
#include <hip/hip_bf16.h>
#include <math.h>

typedef __attribute__((ext_vector_type(16))) _Float16 v16h;
typedef __attribute__((ext_vector_type(8)))  float    v8f;
typedef __attribute__((ext_vector_type(4)))  unsigned int u32x4;
typedef __attribute__((ext_vector_type(4)))  int      i32x4;
typedef __attribute__((ext_vector_type(8)))  int      i32x8;

#define NB    8
#define CCH   256
#define LL    4096
#define BLR   32768   // NB * LL rows
#define DI    64
#define DSN   16

// ---------------------------------------------------------------------------
// Tensor Data Mover: 1-D f32 tile Global -> LDS (D# built per ISA 08 §8).
//   group0: count=1, lds_addr, global_addr[56:0], type=2
//   group1: data_size=4B, tensor_dim0=tile_dim0=nelem (1-D tile)
//   groups 2/3: zero (<=2D tensor)
// ---------------------------------------------------------------------------
__device__ __forceinline__ void tdm_load_f32_1d(unsigned lds_off,
                                                const float* gsrc,
                                                unsigned nelem) {
  unsigned long long ga = (unsigned long long)(uintptr_t)gsrc;
  u32x4 g0;
  g0[0] = 1u;                                                // count=1
  g0[1] = lds_off;                                           // lds_addr (bytes)
  g0[2] = (unsigned)ga;                                      // global_addr[31:0]
  g0[3] = (unsigned)((ga >> 32) & 0x1ffffffu) | (2u << 30);  // [56:32] | type=2
  i32x8 g1;
  g1[0] = (2 << 16);                          // data_size=2 (4 bytes), mask=0
  g1[1] = (int)((nelem & 0xffffu) << 16);     // tensor_dim0[15:0] -> bits[63:48]
  g1[2] = (int)(nelem >> 16);                 // tensor_dim0[31:16] -> bits[79:64]
  g1[3] = (int)((nelem & 0xffffu) << 16);     // tile_dim0 -> bits[127:112]
  g1[4] = 0;                                  // tile_dim1=0, tile_dim2=0 (1-D)
  g1[5] = 0; g1[6] = 0; g1[7] = 0;            // strides unused
  i32x4 z4; z4[0] = 0; z4[1] = 0; z4[2] = 0; z4[3] = 0;
#if __clang_major__ >= 23
  i32x8 z8; z8[0]=0; z8[1]=0; z8[2]=0; z8[3]=0; z8[4]=0; z8[5]=0; z8[6]=0; z8[7]=0;
  __builtin_amdgcn_tensor_load_to_lds(g0, g1, z4, z4, z8, 0);
#else
  __builtin_amdgcn_tensor_load_to_lds(g0, g1, z4, z4, 0);
#endif
}

// ---------------------------------------------------------------------------
// K1: identity + 3 depthwise convs (channels 0..223). Pure bandwidth.
// ---------------------------------------------------------------------------
__global__ __launch_bounds__(256) void k_dwconv(
    const float* __restrict__ x,
    const float* __restrict__ w_hw, const float* __restrict__ b_hw,
    const float* __restrict__ w_w,  const float* __restrict__ b_w,
    const float* __restrict__ w_h,  const float* __restrict__ b_h,
    float* __restrict__ out)
{
  int idx = blockIdx.x * 256 + threadIdx.x;
  const int total = NB * 224 * LL;
  if (idx >= total) return;
  int b  = idx / (224 * LL);
  int r  = idx % (224 * LL);
  int c  = r / LL;
  int hw = r % LL;
  int h = hw >> 6, w = hw & 63;
  const float* xin = x + (size_t)(b * CCH) * LL;
  float v;
  if (c < 128) {
    v = xin[(size_t)c * LL + hw];
  } else if (c < 160) {
    int g = c - 128;
    const float* xc = xin + (size_t)c * LL;
    float acc = b_hw[g];
    for (int ky = 0; ky < 3; ++ky) {
      int h2 = h - 1 + ky;
      if ((unsigned)h2 >= 64u) continue;
      for (int kx = 0; kx < 3; ++kx) {
        int w2 = w - 1 + kx;
        if ((unsigned)w2 >= 64u) continue;
        acc += xc[h2 * 64 + w2] * w_hw[g * 9 + ky * 3 + kx];
      }
    }
    v = acc;
  } else if (c < 192) {
    int g = c - 160;
    const float* xc = xin + (size_t)c * LL;
    float acc = b_w[g];
    for (int kx = 0; kx < 11; ++kx) {
      int w2 = w - 5 + kx;
      if ((unsigned)w2 >= 64u) continue;
      acc += xc[h * 64 + w2] * w_w[g * 11 + kx];
    }
    v = acc;
  } else {
    int g = c - 192;
    const float* xc = xin + (size_t)c * LL;
    float acc = b_h[g];
    for (int ky = 0; ky < 11; ++ky) {
      int h2 = h - 5 + ky;
      if ((unsigned)h2 >= 64u) continue;
      acc += xc[h2 * 64 + w] * w_h[g * 11 + ky];
    }
    v = acc;
  }
  out[((size_t)b * CCH + c) * LL + hw] = v;
}

// ---------------------------------------------------------------------------
// K2: LayerNorm(C=32) + in_proj GEMM [BLR,32]x[32,128] via WMMA f16 (K=32).
// 256 threads = 8 waves, each wave owns 16 rows; block = 128 rows.
// ---------------------------------------------------------------------------
__global__ __launch_bounds__(256) void k_ln_inproj(
    const float* __restrict__ x,
    const float* __restrict__ ln_g, const float* __restrict__ ln_b,
    const float* __restrict__ inw,          // [128][32]
    float* __restrict__ xm, float* __restrict__ zv)
{
  __shared__ _Float16 wts[128 * 32];
  __shared__ _Float16 xn[8][16][32];
  int tid = threadIdx.x;
  int wave = tid >> 5, lane = tid & 31;

  for (int i = tid; i < 128 * 32; i += 256) wts[i] = (_Float16)inw[i];

  int rowbase = blockIdx.x * 128 + wave * 16;
  int b  = rowbase / LL;
  int l0 = rowbase % LL;
  const float* src = x + ((size_t)(b * CCH + 224)) * LL + (size_t)l0 * 32;

  // lane holds row = lane>>1, half = lane&1 (16 contiguous floats)
  float vbuf[16];
  {
    const float* p = src + lane * 16;
    #pragma unroll
    for (int k = 0; k < 16; ++k) vbuf[k] = p[k];
  }
  float s1 = 0.f, s2 = 0.f;
  #pragma unroll
  for (int k = 0; k < 16; ++k) { s1 += vbuf[k]; s2 += vbuf[k] * vbuf[k]; }
  float o1 = __shfl_xor(s1, 1);
  float o2 = __shfl_xor(s2, 1);
  float mu  = (s1 + o1) * (1.f / 32.f);
  float var = (s2 + o2) * (1.f / 32.f) - mu * mu;
  float rstd = rsqrtf(var + 1e-5f);
  int row2 = lane >> 1;
  int cbase = (lane & 1) * 16;
  #pragma unroll
  for (int k = 0; k < 16; ++k) {
    int c = cbase + k;
    float val = (vbuf[k] - mu) * rstd * ln_g[c] + ln_b[c];
    xn[wave][row2][c] = (_Float16)val;
  }
  __syncthreads();

  // A fragment: 16x32 f16 layout
  int rowA = lane & 15;
  int koff = (lane >> 4) * 8;
  v16h a;
  #pragma unroll
  for (int j = 0; j < 16; ++j) {
    int K = ((lane >> 4) << 4) + j;
    a[j] = xn[wave][rowA][K];
  }
  int col = lane & 15;
  int kb  = (lane >> 4) * 16;
  for (int t = 0; t < 8; ++t) {
    v16h bf;
    #pragma unroll
    for (int j = 0; j < 16; ++j) bf[j] = wts[(t * 16 + col) * 32 + kb + j];
    v8f acc;
    #pragma unroll
    for (int i = 0; i < 8; ++i) acc[i] = 0.f;
    acc = __builtin_amdgcn_wmma_f32_16x16x32_f16(false, a, false, bf,
                                                 (short)0, acc, false, false);
    int e = t * 16 + col;
    #pragma unroll
    for (int v = 0; v < 8; ++v) {
      int rr = v + ((lane >> 4) << 3);
      size_t grow = (size_t)(rowbase + rr);
      if (e < 64) xm[grow * 64 + e] = acc[v];
      else        zv[grow * 64 + (e - 64)] = acc[v];
    }
  }
}

// ---------------------------------------------------------------------------
// K3: causal depthwise conv1d (window 4) along L + SiLU -> u
// ---------------------------------------------------------------------------
__global__ __launch_bounds__(256) void k_conv1d(
    const float* __restrict__ xm, const float* __restrict__ cw,
    const float* __restrict__ cb, float* __restrict__ u)
{
  int idx = blockIdx.x * 256 + threadIdx.x;
  if (idx >= BLR * DI) return;
  int d   = idx & 63;
  int row = idx >> 6;
  int l   = row & (LL - 1);
  float acc = cb[d];
  #pragma unroll
  for (int k = 0; k < 4; ++k) {
    int ll = l - 3 + k;
    if (ll >= 0) acc += xm[(size_t)(row - 3 + k) * 64 + d] * cw[d * 4 + k];
  }
  acc = acc / (1.f + __expf(-acc));   // silu
  u[idx] = acc;
}

// ---------------------------------------------------------------------------
// K4: x_proj GEMM [BLR,64]x[64,34] via 2x WMMA (K=64), + dt_proj + softplus.
// Produces delta[BLR,64] and interleaved BCm[BLR][32] (B in 0..15, C in 16..31)
// so the scan kernel can stream one contiguous TDM tile per chunk.
// ---------------------------------------------------------------------------
__global__ __launch_bounds__(256) void k_xproj(
    const float* __restrict__ u,
    const float* __restrict__ xw,    // [34][64]
    const float* __restrict__ dtw,   // [64][2]
    const float* __restrict__ dtb,   // [64]
    float* __restrict__ delta, float* __restrict__ BCm)
{
  __shared__ _Float16 wtx[48 * 64];
  __shared__ _Float16 ub[8][16 * 64];
  __shared__ float dtbuf[8][16][2];
  int tid = threadIdx.x, wave = tid >> 5, lane = tid & 31;

  for (int i = tid; i < 48 * 64; i += 256) {
    int e = i >> 6;
    wtx[i] = (e < 34) ? (_Float16)xw[i] : (_Float16)0.f;
  }
  int rowbase = blockIdx.x * 128 + wave * 16;
  for (int t = 0; t < 32; ++t) {
    int idx = t * 32 + lane;  // 0..1023 covers 16 rows x 64
    ub[wave][idx] = (_Float16)u[(size_t)(rowbase + (idx >> 6)) * 64 + (idx & 63)];
  }
  __syncthreads();

  int rowA = lane & 15;
  int koff = (lane >> 4) * 8;
  int col  = lane & 15;
  int kb   = (lane >> 4) * 16;
  v16h a0, a1;
  #pragma unroll
  for (int j = 0; j < 16; ++j) {
    int K = ((lane >> 4) << 4) + j;
    a0[j] = ub[wave][rowA * 64 + K];
    a1[j] = ub[wave][rowA * 64 + 32 + K];
  }
  for (int t = 0; t < 3; ++t) {
    v16h b0, b1;
    int e = t * 16 + col;
    #pragma unroll
    for (int j = 0; j < 16; ++j) {
      b0[j] = wtx[e * 64 + kb + j];
      b1[j] = wtx[e * 64 + 32 + kb + j];
    }
    v8f acc;
    #pragma unroll
    for (int i = 0; i < 8; ++i) acc[i] = 0.f;
    acc = __builtin_amdgcn_wmma_f32_16x16x32_f16(false, a0, false, b0,
                                                 (short)0, acc, false, false);
    acc = __builtin_amdgcn_wmma_f32_16x16x32_f16(false, a1, false, b1,
                                                 (short)0, acc, false, false);
    #pragma unroll
    for (int v = 0; v < 8; ++v) {
      int rr = v + ((lane >> 4) << 3);
      size_t grow = (size_t)(rowbase + rr);
      float val = acc[v];
      if (e < 2)       dtbuf[wave][rr][e] = val;
      else if (e < 34) BCm[grow * 32 + (e - 2)] = val;   // B: n<16, C: n>=16
    }
  }
  __syncthreads();

  // delta = softplus(dt @ dt_proj_w^T + dt_proj_b), 16 rows x 64 d per wave
  float dw0a = dtw[lane * 2], dw1a = dtw[lane * 2 + 1], dba = dtb[lane];
  float dw0b = dtw[(lane + 32) * 2], dw1b = dtw[(lane + 32) * 2 + 1], dbb = dtb[lane + 32];
  for (int rr = 0; rr < 16; ++rr) {
    float t0 = dtbuf[wave][rr][0], t1 = dtbuf[wave][rr][1];
    size_t grow = (size_t)(rowbase + rr);
    float xa = t0 * dw0a + t1 * dw1a + dba;
    float xb = t0 * dw0b + t1 * dw1b + dbb;
    delta[grow * 64 + lane]      = (xa > 20.f) ? xa : log1pf(__expf(xa));
    delta[grow * 64 + lane + 32] = (xb > 20.f) ? xb : log1pf(__expf(xb));
  }
}

// ---------------------------------------------------------------------------
// K5: selective scan. One block per batch b, 64 threads = one d each.
// 16 states in registers; interleaved B/C vectors double-buffered into LDS
// via the Tensor Data Mover (8KB tile per 64-step chunk), overlapped with the
// latency-bound recurrence. Fuses y = (y + u*D) * silu(z).
// ---------------------------------------------------------------------------
__global__ __launch_bounds__(64) void k_scan(
    const float* __restrict__ u, const float* __restrict__ delta,
    const float* __restrict__ BCm, const float* __restrict__ zv,
    const float* __restrict__ A_log, const float* __restrict__ Dv,
    float* __restrict__ yf)
{
  __shared__ float sBC[2][64 * 32];
  int b = blockIdx.x;
  int d = threadIdx.x;
  float a[DSN], h[DSN];
  #pragma unroll
  for (int n = 0; n < DSN; ++n) { a[n] = -__expf(A_log[d * DSN + n]); h[n] = 0.f; }
  float Dd = Dv[d];
  size_t base = (size_t)b * LL;
  bool issuer = (threadIdx.x < 32);   // wave 0 drives the TDM

  if (issuer) {
    tdm_load_f32_1d((unsigned)(uintptr_t)&sBC[0][0], &BCm[base * 32], 64 * 32);
    __builtin_amdgcn_s_wait_tensorcnt(0);
  }
  __syncthreads();

  int buf = 0;
  for (int c0 = 0; c0 < LL; c0 += 64) {
    if (issuer && (c0 + 64 < LL))
      tdm_load_f32_1d((unsigned)(uintptr_t)&sBC[buf ^ 1][0],
                      &BCm[(base + c0 + 64) * 32], 64 * 32);
    if (c0 + 64 < LL)
      __builtin_prefetch(&delta[(base + c0 + 64) * 64], 0, 1);

    const float* bc = sBC[buf];
    for (int i = 0; i < 64; ++i) {
      size_t row = base + c0 + i;
      float dl = delta[row * 64 + d];
      float uu = u[row * 64 + d];
      float du = dl * uu;
      float y = 0.f;
      #pragma unroll
      for (int n = 0; n < DSN; ++n) {
        h[n] = h[n] * __expf(dl * a[n]) + du * bc[i * 32 + n];
        y += h[n] * bc[i * 32 + 16 + n];
      }
      float z = zv[row * 64 + d];
      yf[row * 64 + d] = (y + uu * Dd) * (z / (1.f + __expf(-z)));
    }
    if (issuer) __builtin_amdgcn_s_wait_tensorcnt(0);
    __syncthreads();
    buf ^= 1;
  }
}

// ---------------------------------------------------------------------------
// K6: out_proj GEMM [BLR,64]x[64,32] via 2x WMMA, write mamba output channels
// (raw-reshape layout: out[b][224 + j/4096][j%4096], j = l*32 + e).
// ---------------------------------------------------------------------------
__global__ __launch_bounds__(256) void k_outproj(
    const float* __restrict__ yf, const float* __restrict__ ow,  // [32][64]
    float* __restrict__ out)
{
  __shared__ _Float16 wo[32 * 64];
  __shared__ _Float16 yb[8][16 * 64];
  int tid = threadIdx.x, wave = tid >> 5, lane = tid & 31;
  for (int i = tid; i < 32 * 64; i += 256) wo[i] = (_Float16)ow[i];
  int rowbase = blockIdx.x * 128 + wave * 16;
  for (int t = 0; t < 32; ++t) {
    int idx = t * 32 + lane;
    yb[wave][idx] = (_Float16)yf[(size_t)(rowbase + (idx >> 6)) * 64 + (idx & 63)];
  }
  __syncthreads();

  int rowA = lane & 15;
  int koff = (lane >> 4) * 8;
  int col  = lane & 15;
  int kb   = (lane >> 4) * 16;
  v16h a0, a1;
  #pragma unroll
  for (int j = 0; j < 16; ++j) {
    int K = ((lane >> 4) << 4) + j;
    a0[j] = yb[wave][rowA * 64 + K];
    a1[j] = yb[wave][rowA * 64 + 32 + K];
  }
  for (int t = 0; t < 2; ++t) {
    v16h b0, b1;
    int e = t * 16 + col;
    #pragma unroll
    for (int j = 0; j < 16; ++j) {
      b0[j] = wo[e * 64 + kb + j];
      b1[j] = wo[e * 64 + 32 + kb + j];
    }
    v8f acc;
    #pragma unroll
    for (int i = 0; i < 8; ++i) acc[i] = 0.f;
    acc = __builtin_amdgcn_wmma_f32_16x16x32_f16(false, a0, false, b0,
                                                 (short)0, acc, false, false);
    acc = __builtin_amdgcn_wmma_f32_16x16x32_f16(false, a1, false, b1,
                                                 (short)0, acc, false, false);
    #pragma unroll
    for (int v = 0; v < 8; ++v) {
      int grow = rowbase + v + ((lane >> 4) << 3);
      int bb = grow >> 12;        // /4096
      int l  = grow & 4095;
      out[((size_t)bb * CCH + 224) * LL + (size_t)l * 32 + e] = acc[v];
    }
  }
}

// ---------------------------------------------------------------------------
extern "C" void kernel_launch(void* const* d_in, const int* in_sizes, int n_in,
                              void* d_out, int out_size, void* d_ws, size_t ws_size,
                              hipStream_t stream) {
  const float* x    = (const float*)d_in[0];
  const float* w_hw = (const float*)d_in[1];
  const float* b_hw = (const float*)d_in[2];
  const float* w_w  = (const float*)d_in[3];
  const float* b_w  = (const float*)d_in[4];
  const float* w_h  = (const float*)d_in[5];
  const float* b_h  = (const float*)d_in[6];
  const float* ln_g = (const float*)d_in[7];
  const float* ln_b = (const float*)d_in[8];
  const float* inw  = (const float*)d_in[9];
  const float* cw   = (const float*)d_in[10];
  const float* cb   = (const float*)d_in[11];
  const float* xw   = (const float*)d_in[12];
  const float* dtw  = (const float*)d_in[13];
  const float* dtb  = (const float*)d_in[14];
  const float* alog = (const float*)d_in[15];
  const float* Dv   = (const float*)d_in[16];
  const float* ow   = (const float*)d_in[17];
  float* out = (float*)d_out;

  float* w    = (float*)d_ws;
  float* xm   = w;                              // [BLR][64] (reused as yf)
  float* zv   = w + (size_t)1 * BLR * 64;       // [BLR][64]
  float* uu   = w + (size_t)2 * BLR * 64;       // [BLR][64]
  float* dl   = w + (size_t)3 * BLR * 64;       // [BLR][64]
  float* BCm  = w + (size_t)4 * BLR * 64;       // [BLR][32] interleaved B|C
  float* yf   = xm;

  k_dwconv  <<<(NB * 224 * LL) / 256, 256, 0, stream>>>(x, w_hw, b_hw, w_w, b_w, w_h, b_h, out);
  k_ln_inproj<<<BLR / 128, 256, 0, stream>>>(x, ln_g, ln_b, inw, xm, zv);
  k_conv1d  <<<(BLR * DI) / 256, 256, 0, stream>>>(xm, cw, cb, uu);
  k_xproj   <<<BLR / 128, 256, 0, stream>>>(uu, xw, dtw, dtb, dl, BCm);
  k_scan    <<<NB, 64, 0, stream>>>(uu, dl, BCm, zv, alog, Dv, yf);
  k_outproj <<<BLR / 128, 256, 0, stream>>>(yf, ow, out);
}
